// MemTransformerLM_12850542150555
// MI455X (gfx1250) — compile-verified, matrix operand order checked by reading, not run on hardware
//
#include <hip/hip_runtime.h>
#include <hip/hip_bf16.h>

// ---------------------------------------------------------------------------
// Transformer-XL layer on MI455X (gfx1250):
//   bf16 WMMA + TDM (tensor_load_to_lds) GEMM staging + async global->LDS attn.
// QLEN=512 MLEN=512 KLEN=1024 BSZ=8 DM=1024 NH=16 DH=64 DI=4096
// ---------------------------------------------------------------------------

#define QLEN 512
#define MLEN 512
#define KLEN 1024
#define BSZ  8
#define DM   1024
#define NH   16
#define DH   64
#define DI   4096
#define HD   (NH * DH)   // 1024

typedef __bf16 bf16_t;
typedef __attribute__((ext_vector_type(16))) __bf16 bf16x16;
typedef __attribute__((ext_vector_type(8)))  float  floatx8;
typedef __attribute__((ext_vector_type(4)))  unsigned int u32x4;
typedef __attribute__((ext_vector_type(8)))  unsigned int u32x8;

union Frag {
  bf16x16 v;
  uint4   q[2];
  __bf16  h[16];
};

__device__ __forceinline__ floatx8 wmma_bf16(const Frag& a, const Frag& b, floatx8 c) {
  // 8 args: (neg_a, A, neg_b, B, c_mod, C, reuse_a, reuse_b)
  return __builtin_amdgcn_wmma_f32_16x16x32_bf16(false, a.v, false, b.v,
                                                 (short)0, c, false, false);
}

// ---------------------------------------------------------------------------
// CDNA5 async copy: 16B global -> LDS, no VGPR round-trip, tracked by ASYNCcnt.
// VDST operand carries the LDS byte address (low 32 bits of the flat pointer
// to __shared__ memory per ISA 10.2 LDS aperture construction).
// ---------------------------------------------------------------------------
__device__ __forceinline__ void async_b128(void* lds_dst, const void* gsrc) {
  unsigned laddr = (unsigned)(size_t)lds_dst;
  asm volatile("global_load_async_to_lds_b128 %0, %1, off"
               :: "v"(laddr), "v"(gsrc)
               : "memory");
}

__device__ __forceinline__ void wait_async0() {
  asm volatile("s_wait_asynccnt 0x0" ::: "memory");
}

// ---------------------------------------------------------------------------
// CDNA5 Tensor Data Mover: DMA one [128 rows x 32 bf16] tile, row stride
// `row_stride_elems`, straight into LDS with the hardware applying our
// 8-bf16 (4-DWORD) row padding every 64B. 2D descriptor => groups 2/3 NULL.
// D# fields per CDNA5 ISA:
//   g0: [1:0]=count=1, [63:32]=lds_addr, [120:64]=global_addr, [127:126]=2
//   g1: [17:16]=data_size(1=2B), [20]=pad_enable, [24:22]=pad_interval(3=16dw),
//       [31:25]=pad_amount(3=4dw), dims huge (no OOB zero-fill),
//       [127:112]=tile_dim0=32, [143:128]=tile_dim1=128, [207:160]=dim0 stride.
// Issued by one wave per workgroup; completion via s_wait_tensorcnt.
// ---------------------------------------------------------------------------
__device__ __forceinline__ void tdm_load_tile(void* lds_dst, const void* gsrc,
                                              unsigned row_stride_elems) {
  const unsigned long long ga = (unsigned long long)(size_t)gsrc;
  u32x4 g0;
  g0[0] = 1u;                                    // count=1 (valid), user mode
  g0[1] = (unsigned)(size_t)lds_dst;             // lds_addr (bytes)
  g0[2] = (unsigned)ga;                          // global_addr[31:0]
  g0[3] = ((unsigned)(ga >> 32) & 0x01FFFFFFu)   // global_addr[56:32]
          | (2u << 30);                          // type = 2 ("image")
  u32x8 g1;
  g1[0] = (1u << 16)                             // data_size = 1 -> 2 bytes
        | (1u << 20)                             // pad_enable
        | (3u << 22)                             // pad_interval: 16 DWORDs (64B)
        | (3u << 25);                            // pad_amount: 4 DWORDs (8 bf16)
  g1[1] = 0xFFFFu << 16;                         // tensor_dim0[15:0] of 0x7FFFFFFF
  g1[2] = 0x7FFFu | (0xFFFFu << 16);             // tensor_dim0[31:16] | tensor_dim1[15:0]
  g1[3] = 0x7FFFu | (32u << 16);                 // tensor_dim1[31:16] | tile_dim0=32
  g1[4] = 128u;                                  // tile_dim1=128, tile_dim2=0
  g1[5] = row_stride_elems;                      // tensor_dim0_stride[31:0]
  g1[6] = 0u;                                    // stride[47:32]=0 | dim1_stride lo
  g1[7] = 0u;                                    // dim1_stride hi (unused, 2D)
  asm volatile("tensor_load_to_lds %0, %1" :: "s"(g0), "s"(g1) : "memory");
}

// ---------------------------------------------------------------------------
// fp32 -> bf16 straight copy
// ---------------------------------------------------------------------------
__global__ __launch_bounds__(256) void k_cvt(const float* __restrict__ src,
                                             bf16_t* __restrict__ dst, int n) {
  int i = blockIdx.x * 256 + threadIdx.x;
  if (i < n) dst[i] = (bf16_t)src[i];
}

// ---------------------------------------------------------------------------
// fp32 [R][C] -> bf16 [C][R] (transpose + convert), R,C multiples of 32
// ---------------------------------------------------------------------------
__global__ __launch_bounds__(256) void k_tcvt(const float* __restrict__ src,
                                              bf16_t* __restrict__ dst,
                                              int R, int C) {
  __shared__ float t[32][33];
  int rb = blockIdx.y * 32, cb = blockIdx.x * 32;
  int tx = threadIdx.x & 31, ty = threadIdx.x >> 5;   // 32 x 8
  for (int i = ty; i < 32; i += 8)
    t[i][tx] = src[(size_t)(rb + i) * C + cb + tx];
  __syncthreads();
  for (int i = ty; i < 32; i += 8)
    dst[(size_t)(cb + i) * R + rb + tx] = (bf16_t)t[tx][i];
}

// ---------------------------------------------------------------------------
// Generic NT GEMM: C[M x N] = A[M x K] * Bt[N x K]^T   (bf16 in, f32 accum)
// Block tile 128x128 (256 threads = 8 waves, each wave 64x32, K-step 32).
// LDS tiles are double-buffered; the Tensor Data Mover streams tile k+1 into
// buf^1 (one tensor_load_to_lds per matrix, hardware-padded rows) while the
// WMMAs consume buf. M,N multiples of 128; K multiple of 32.
// ---------------------------------------------------------------------------
__global__ __launch_bounds__(256) void k_gemm(
    const bf16_t* __restrict__ A, const bf16_t* __restrict__ Bt,
    int M, int N, int K,
    const float* __restrict__ bias, const float* __restrict__ resid, int do_relu,
    float* __restrict__ outF, bf16_t* __restrict__ outB)
{
  __shared__ __align__(16) bf16_t As[2][128][40];   // padded rows: 80B, 16B aligned
  __shared__ __align__(16) bf16_t Bs[2][128][40];

  const int tid  = threadIdx.x;
  const int lane = tid & 31;
  const int w    = tid >> 5;
  const int wm   = w >> 2;        // 0..1  (rows of 64)
  const int wn   = w & 3;         // 0..3  (cols of 32)
  const int half = lane >> 4;     // K-half select / M-half in C layout
  const int l15  = lane & 15;
  const size_t rowBase = (size_t)blockIdx.y * 128;
  const size_t colBase = (size_t)blockIdx.x * 128;
  const bf16_t* Atile = A  + rowBase * K;
  const bf16_t* Btile = Bt + colBase * K;

  floatx8 acc[4][2] = {};

  // prologue: TDM-stage tile 0 into buffer 0 (wave 0 drives the DMA)
  if (w == 0) {
    tdm_load_tile(&As[0][0][0], Atile, (unsigned)K);
    tdm_load_tile(&Bs[0][0][0], Btile, (unsigned)K);
    __builtin_amdgcn_s_wait_tensorcnt(0);
  }
  __syncthreads();

  int buf = 0;
  for (int k0 = 0; k0 < K; k0 += 32) {
    // ---- TDM-stage next K tile into the other buffer ----
    if (w == 0 && k0 + 32 < K) {
      tdm_load_tile(&As[buf ^ 1][0][0], Atile + k0 + 32, (unsigned)K);
      tdm_load_tile(&Bs[buf ^ 1][0][0], Btile + k0 + 32, (unsigned)K);
    }

    // ---- fragments (CDNA5 16-bit A layout: lanes<16 K{0-7,16-23}, else K{8-15,24-31}) ----
    Frag af[4], bfg[2];
#pragma unroll
    for (int tm = 0; tm < 4; ++tm) {
      const uint4* ar = (const uint4*)&As[buf][wm * 64 + tm * 16 + l15][0];
      af[tm].q[0] = ar[half];
      af[tm].q[1] = ar[2 + half];
    }
#pragma unroll
    for (int tn = 0; tn < 2; ++tn) {
      const uint4* br = (const uint4*)&Bs[buf][wn * 32 + tn * 16 + l15][0];
      bfg[tn].q[0] = br[half];
      bfg[tn].q[1] = br[2 + half];
    }
#pragma unroll
    for (int tm = 0; tm < 4; ++tm)
#pragma unroll
      for (int tn = 0; tn < 2; ++tn)
        acc[tm][tn] = wmma_bf16(af[tm], bfg[tn], acc[tm][tn]);

    if (w == 0) __builtin_amdgcn_s_wait_tensorcnt(0);  // next tile landed
    __syncthreads();                                   // all waves done with buf
    buf ^= 1;
  }

  // ---- epilogue (C layout: row = vgpr r + 8*(lane>=16), col = lane&15) ----
#pragma unroll
  for (int tm = 0; tm < 4; ++tm) {
#pragma unroll
    for (int tn = 0; tn < 2; ++tn) {
#pragma unroll
      for (int r = 0; r < 8; ++r) {
        size_t row = rowBase + wm * 64 + tm * 16 + r + 8 * half;
        size_t col = colBase + wn * 32 + tn * 16 + l15;
        float v = acc[tm][tn][r];
        if (bias)    v += bias[col];
        if (do_relu) v  = fmaxf(v, 0.0f);
        if (resid)   v += resid[row * N + col];
        if (outF) outF[row * N + col] = v;
        if (outB) outB[row * N + col] = (bf16_t)v;
      }
    }
  }
}

// ---------------------------------------------------------------------------
// LayerNorm over rows of 1024 f32, one 256-thread block per row.
// ---------------------------------------------------------------------------
__global__ __launch_bounds__(256) void k_ln(const float* __restrict__ x,
                                            const float* __restrict__ g,
                                            const float* __restrict__ bta,
                                            float* __restrict__ outF,
                                            bf16_t* __restrict__ outB)
{
  __shared__ float red[256];
  const int t = threadIdx.x;
  const size_t row = blockIdx.x;
  const float4 v = ((const float4*)(x + row * DM))[t];
  red[t] = v.x + v.y + v.z + v.w;
  __syncthreads();
  for (int o = 128; o > 0; o >>= 1) { if (t < o) red[t] += red[t + o]; __syncthreads(); }
  const float mu = red[0] * (1.0f / DM);
  __syncthreads();
  const float d0 = v.x - mu, d1 = v.y - mu, d2 = v.z - mu, d3 = v.w - mu;
  red[t] = d0 * d0 + d1 * d1 + d2 * d2 + d3 * d3;
  __syncthreads();
  for (int o = 128; o > 0; o >>= 1) { if (t < o) red[t] += red[t + o]; __syncthreads(); }
  const float rstd = rsqrtf(red[0] * (1.0f / DM) + 1e-5f);
  const float dv[4] = {d0, d1, d2, d3};
#pragma unroll
  for (int j = 0; j < 4; ++j) {
    int c = t * 4 + j;
    float y = dv[j] * rstd * g[c] + bta[c];
    if (outF) outF[row * DM + c] = y;
    if (outB) outB[row * DM + c] = (bf16_t)y;
  }
}

// ---------------------------------------------------------------------------
// Fused attention: one block per (32 q-rows, batch b, head h).
// Full 32x1024 score strip lives in LDS (~165 KB dynamic LDS; CDNA5: 320KB/WGP).
//   AC  = (Q + r_w_bias) . K^T        (WMMA, K tile async-staged)
//   BDr = (Q + r_r_bias) . Rk^T      (WMMA, Rk tile async-staged),
//         rel-shift scatter: j = p - 511 + i_global
//   mask j > i+MLEN, scale 1/8, row softmax, then P.V (WMMA).
// ---------------------------------------------------------------------------
#define AROWS 32

__global__ __launch_bounds__(256) void k_attn(
    const bf16_t* __restrict__ heads,   // [KLEN*BSZ][3*HD], q|k|v blocks of 1024 cols
    const bf16_t* __restrict__ rk,      // [KLEN][HD]
    const float*  __restrict__ rwb,     // [NH*DH]
    const float*  __restrict__ rrb,     // [NH*DH]
    bf16_t* __restrict__ vecb)          // [QLEN*BSZ][HD]
{
  extern __shared__ char smem[];
  float*  S   = (float*)smem;                          // [32][1024] f32   131072B
  bf16_t* Qw  = (bf16_t*)(smem + AROWS * KLEN * 4);    // [32][72]         4608B
  bf16_t* Qr  = Qw  + AROWS * 72;                      // [32][72]
  bf16_t* Kt  = Qr  + AROWS * 72;                      // [64][72]
  bf16_t* Rkt = Kt  + 64 * 72;                         // [64][72]
  bf16_t* Vt  = Rkt + 64 * 72;                         // [64][40]  (V^T tile)
  float*  red = (float*)(Vt + 64 * 40);                // [32][8]

  const int tid  = threadIdx.x;
  const int lane = tid & 31;
  const int w    = tid >> 5;
  const int wm   = w >> 2, wn = w & 3;                 // 2 x 4 wave grid, 16x16 tiles
  const int half = lane >> 4, l15 = lane & 15;
  const int i0   = blockIdx.x * AROWS;
  const int b    = blockIdx.y;
  const int h    = blockIdx.z;
  const size_t strideH = 3 * HD;

  // ---- stage Q twice, with the two biases folded in (scalar: needs ALU) ----
  for (int e = tid; e < AROWS * DH; e += 256) {
    int i = e >> 6, d = e & 63;
    size_t grow = (size_t)(MLEN + i0 + i) * BSZ + b;
    float qv = (float)heads[grow * strideH + h * DH + d];
    Qw[i * 72 + d] = (bf16_t)(qv + rwb[h * DH + d]);
    Qr[i * 72 + d] = (bf16_t)(qv + rrb[h * DH + d]);
  }
  __syncthreads();

  // ---- AC phase: S[i][j] = Qw . K^T ----
  for (int jt = 0; jt < KLEN / 64; ++jt) {
    // K tile: 64 rows x 64 bf16 = 512 x 16B chunks, streamed straight to LDS
    for (int e = tid; e < 64 * 8; e += 256) {
      int j = e >> 3, c = (e & 7) << 3;
      size_t grow = (size_t)(jt * 64 + j) * BSZ + b;
      async_b128(&Kt[j * 72 + c], heads + grow * strideH + HD + h * DH + c);
    }
    wait_async0();
    __syncthreads();
    Frag a0, a1, b0, b1;
    const uint4* ar = (const uint4*)&Qw[(wm * 16 + l15) * 72];
    a0.q[0] = ar[half];     a0.q[1] = ar[2 + half];        // K 0..31
    a1.q[0] = ar[4 + half]; a1.q[1] = ar[6 + half];        // K 32..63
    const uint4* br = (const uint4*)&Kt[(wn * 16 + l15) * 72];
    b0.q[0] = br[half];     b0.q[1] = br[2 + half];
    b1.q[0] = br[4 + half]; b1.q[1] = br[6 + half];
    floatx8 acc = {};
    acc = wmma_bf16(a0, b0, acc);
    acc = wmma_bf16(a1, b1, acc);
#pragma unroll
    for (int r = 0; r < 8; ++r) {
      int i = wm * 16 + r + 8 * half;
      int j = jt * 64 + wn * 16 + l15;
      S[i * KLEN + j] = acc[r];
    }
    __syncthreads();
  }

  // ---- BD phase: BDraw[i][p] = Qr . Rk^T, scatter to S[i][p - 511 + i_glob] ----
  for (int pt = 0; pt < KLEN / 64; ++pt) {
    for (int e = tid; e < 64 * 8; e += 256) {
      int j = e >> 3, c = (e & 7) << 3;
      async_b128(&Rkt[j * 72 + c], rk + (size_t)(pt * 64 + j) * HD + h * DH + c);
    }
    wait_async0();
    __syncthreads();
    Frag a0, a1, b0, b1;
    const uint4* ar = (const uint4*)&Qr[(wm * 16 + l15) * 72];
    a0.q[0] = ar[half];     a0.q[1] = ar[2 + half];
    a1.q[0] = ar[4 + half]; a1.q[1] = ar[6 + half];
    const uint4* br = (const uint4*)&Rkt[(wn * 16 + l15) * 72];
    b0.q[0] = br[half];     b0.q[1] = br[2 + half];
    b1.q[0] = br[4 + half]; b1.q[1] = br[6 + half];
    floatx8 acc = {};
    acc = wmma_bf16(a0, b0, acc);
    acc = wmma_bf16(a1, b1, acc);
#pragma unroll
    for (int r = 0; r < 8; ++r) {
      int i = wm * 16 + r + 8 * half;
      int p = pt * 64 + wn * 16 + l15;
      int j = p - (QLEN - 1) + (i0 + i);          // exact rel_shift mapping
      if ((unsigned)j < (unsigned)KLEN)
        S[i * KLEN + j] += acc[r];                // (i,j) touched exactly once
    }
    __syncthreads();
  }

  // ---- mask + scale ----
  for (int e = tid; e < AROWS * KLEN; e += 256) {
    int i = e >> 10, j = e & (KLEN - 1);
    float v = S[e];
    S[e] = (j <= i0 + i + MLEN) ? v * 0.125f : -1.0e30f;
  }
  __syncthreads();

  // ---- row softmax over 1024 (8 threads per row) ----
  {
    int rrow = tid >> 3, part = tid & 7;
    float* srow = &S[rrow * KLEN];
    float mx = -3.0e38f;
    for (int c = part * 128; c < part * 128 + 128; ++c) mx = fmaxf(mx, srow[c]);
    red[rrow * 8 + part] = mx;
    __syncthreads();
    float m = red[rrow * 8];
#pragma unroll
    for (int q = 1; q < 8; ++q) m = fmaxf(m, red[rrow * 8 + q]);
    float sum = 0.0f;
    for (int c = part * 128; c < part * 128 + 128; ++c) {
      float ev = __expf(srow[c] - m);
      srow[c] = ev;
      sum += ev;
    }
    __syncthreads();
    red[rrow * 8 + part] = sum;
    __syncthreads();
    float tot = 0.0f;
#pragma unroll
    for (int q = 0; q < 8; ++q) tot += red[rrow * 8 + q];
    float inv = 1.0f / tot;
    for (int c = part * 128; c < part * 128 + 128; ++c) srow[c] *= inv;
  }
  __syncthreads();

  // ---- PV: vec[32x64] = P[32x1024] . V[1024x64]  (V staged transposed) ----
  floatx8 vac = {};
  for (int k0 = 0; k0 < KLEN; k0 += 32) {
    for (int e = tid; e < DH * 32; e += 256) {
      int d = e >> 5, jj = e & 31;
      size_t grow = (size_t)(k0 + jj) * BSZ + b;
      Vt[d * 40 + jj] = heads[grow * strideH + 2 * HD + h * DH + d];
    }
    __syncthreads();
    Frag pa, vb;
    {
      const float* prow = &S[(wm * 16 + l15) * KLEN + k0];
      const int c0 = half * 8, c1 = 16 + half * 8;
#pragma unroll
      for (int e2 = 0; e2 < 8; ++e2) {            // f32 probs -> bf16 fragment
        pa.h[e2]     = (bf16_t)prow[c0 + e2];
        pa.h[8 + e2] = (bf16_t)prow[c1 + e2];
      }
    }
    {
      const uint4* vr = (const uint4*)&Vt[(wn * 16 + l15) * 40];
      vb.q[0] = vr[half];
      vb.q[1] = vr[2 + half];
    }
    vac = wmma_bf16(pa, vb, vac);
    __syncthreads();
  }
#pragma unroll
  for (int r = 0; r < 8; ++r) {
    int i = wm * 16 + r + 8 * half;
    int d = wn * 16 + l15;
    size_t grow = (size_t)(i0 + i) * BSZ + b;     // vec row = i*BSZ + b
    vecb[grow * HD + h * DH + d] = (bf16_t)vac[r];
  }
}

// ---------------------------------------------------------------------------
// Host orchestration
// ---------------------------------------------------------------------------
extern "C" void kernel_launch(void* const* d_in, const int* in_sizes, int n_in,
                              void* d_out, int out_size, void* d_ws, size_t ws_size,
                              hipStream_t stream) {
  const float* w     = (const float*)d_in[0];   // [512,8,1024]
  const float* r     = (const float*)d_in[1];   // [1024,1024]
  const float* mems  = (const float*)d_in[2];   // [512,8,1024]
  const float* w_qkv = (const float*)d_in[3];   // [1024,3072]
  const float* w_r   = (const float*)d_in[4];   // [1024,1024]
  const float* w_o   = (const float*)d_in[5];   // [1024,1024]
  const float* rwb   = (const float*)d_in[6];   // [16,64]
  const float* rrb   = (const float*)d_in[7];   // [16,64]
  const float* ln1g  = (const float*)d_in[8];
  const float* ln1b  = (const float*)d_in[9];
  const float* ffw1  = (const float*)d_in[10];  // [1024,4096]
  const float* ffb1  = (const float*)d_in[11];
  const float* ffw2  = (const float*)d_in[12];  // [4096,1024]
  const float* ffb2  = (const float*)d_in[13];
  const float* ln2g  = (const float*)d_in[14];
  const float* ln2b  = (const float*)d_in[15];
  (void)in_sizes; (void)n_in; (void)out_size; (void)ws_size;

  char* ws = (char*)d_ws;
  size_t off = 0;
  auto alloc = [&](size_t bytes) -> void* {
    void* p = ws + off;
    off += (bytes + 255) & ~(size_t)255;
    return p;
  };
  bf16_t* catb   = (bf16_t*)alloc((size_t)KLEN * BSZ * DM * 2);     // cat(mems,w) bf16
  bf16_t* wqkvT  = (bf16_t*)alloc((size_t)3 * HD * DM * 2);         // w_qkv^T
  bf16_t* wrT    = (bf16_t*)alloc((size_t)HD * DM * 2);             // w_r^T
  bf16_t* woT    = (bf16_t*)alloc((size_t)DM * HD * 2);             // w_o^T
  bf16_t* fw1T   = (bf16_t*)alloc((size_t)DI * DM * 2);             // ff_w1^T
  bf16_t* fw2T   = (bf16_t*)alloc((size_t)DM * DI * 2);             // ff_w2^T
  bf16_t* rbf    = (bf16_t*)alloc((size_t)KLEN * DM * 2);           // r bf16
  bf16_t* headsb = (bf16_t*)alloc((size_t)KLEN * BSZ * 3 * HD * 2); // qkv heads bf16
  bf16_t* rkb    = (bf16_t*)alloc((size_t)KLEN * HD * 2);           // r_head_k bf16
  bf16_t* vecb   = (bf16_t*)alloc((size_t)QLEN * BSZ * HD * 2);     // attn vec bf16
  float*  x1     = (float*) alloc((size_t)QLEN * BSZ * DM * 4);     // w + attn_out
  float*  hbuf   = (float*) alloc((size_t)QLEN * BSZ * DM * 4);     // LN1 out f32
  bf16_t* hb     = (bf16_t*)alloc((size_t)QLEN * BSZ * DM * 2);     // LN1 out bf16
  bf16_t* relub  = (bf16_t*)alloc((size_t)QLEN * BSZ * DI * 2);     // relu(ff1) bf16
  float*  x2     = (float*) alloc((size_t)QLEN * BSZ * DM * 4);     // h + ff

  // 1) convert activations to bf16
  const int nMem = MLEN * BSZ * DM, nW = QLEN * BSZ * DM, nR = KLEN * DM;
  k_cvt<<<dim3((nMem + 255) / 256), dim3(256), 0, stream>>>(mems, catb, nMem);
  k_cvt<<<dim3((nW + 255) / 256), dim3(256), 0, stream>>>(w, catb + (size_t)MLEN * BSZ * DM, nW);
  k_cvt<<<dim3((nR + 255) / 256), dim3(256), 0, stream>>>(r, rbf, nR);

  // 2) transpose+convert weights (so every GEMM is K-contiguous NT)
  k_tcvt<<<dim3(3 * HD / 32, DM / 32), dim3(256), 0, stream>>>(w_qkv, wqkvT, DM, 3 * HD);
  k_tcvt<<<dim3(HD / 32, DM / 32), dim3(256), 0, stream>>>(w_r, wrT, DM, HD);
  k_tcvt<<<dim3(DM / 32, HD / 32), dim3(256), 0, stream>>>(w_o, woT, HD, DM);
  k_tcvt<<<dim3(DI / 32, DM / 32), dim3(256), 0, stream>>>(ffw1, fw1T, DM, DI);
  k_tcvt<<<dim3(DM / 32, DI / 32), dim3(256), 0, stream>>>(ffw2, fw2T, DI, DM);

  // 3) qkv heads: [8192 x 3072] = catb[8192x1024] x wqkvT^T
  k_gemm<<<dim3(3 * HD / 128, KLEN * BSZ / 128), dim3(256), 0, stream>>>(
      catb, wqkvT, KLEN * BSZ, 3 * HD, DM,
      (const float*)nullptr, (const float*)nullptr, 0, (float*)nullptr, headsb);

  // 4) r_head_k: [1024 x 1024]
  k_gemm<<<dim3(HD / 128, KLEN / 128), dim3(256), 0, stream>>>(
      rbf, wrT, KLEN, HD, DM,
      (const float*)nullptr, (const float*)nullptr, 0, (float*)nullptr, rkb);

  // 5) fused attention (AC + rel-shifted BD + mask + softmax + PV)
  const size_t smemBytes = (size_t)AROWS * KLEN * 4 + 2 * AROWS * 72 * 2 +
                           2 * 64 * 72 * 2 + 64 * 40 * 2 + AROWS * 8 * 4; // 164864
  k_attn<<<dim3(QLEN / AROWS, BSZ, NH), dim3(256), smemBytes, stream>>>(
      headsb, rkb, rwb, rrb, vecb);

  // 6) out projection + residual w -> x1
  k_gemm<<<dim3(DM / 128, QLEN * BSZ / 128), dim3(256), 0, stream>>>(
      vecb, woT, QLEN * BSZ, DM, HD,
      (const float*)nullptr, w, 0, x1, (bf16_t*)nullptr);

  // 7) LN1 -> h (f32 for residual) + hb (bf16 for FF1)
  k_ln<<<dim3(QLEN * BSZ), dim3(256), 0, stream>>>(x1, ln1g, ln1b, hbuf, hb);

  // 8) FF1 + bias + ReLU -> bf16
  k_gemm<<<dim3(DI / 128, QLEN * BSZ / 128), dim3(256), 0, stream>>>(
      hb, fw1T, QLEN * BSZ, DI, DM,
      ffb1, (const float*)nullptr, 1, (float*)nullptr, relub);

  // 9) FF2 + bias + residual h -> x2
  k_gemm<<<dim3(DM / 128, QLEN * BSZ / 128), dim3(256), 0, stream>>>(
      relub, fw2T, QLEN * BSZ, DM, DI,
      ffb2, hbuf, 0, x2, (bf16_t*)nullptr);

  // 10) LN2 -> output (f32)
  k_ln<<<dim3(QLEN * BSZ), dim3(256), 0, stream>>>(x2, ln2g, ln2b, (float*)d_out,
                                                   (bf16_t*)nullptr);
}
